// Attention_84026740179215
// MI455X (gfx1250) — matrix operand order repaired and checked by reading, not compiled, not run on hardware
//
#include <hip/hip_runtime.h>
#include <hip/hip_bf16.h>

typedef unsigned short u16;
typedef __attribute__((ext_vector_type(16))) __bf16 v16bf;
typedef __attribute__((ext_vector_type(8)))  float  v8f;
typedef __attribute__((ext_vector_type(4)))  int    v4i;

struct alignas(16) U128 { unsigned int x[4]; };
union FragAB { U128 q[2]; v16bf v; };
union V128   { U128 u; u16 s[8]; };

#define SEQ    1024
#define CDIM   768
#define NHEAD  12
#define HDIM   64
#define BATCH  8
#define MTOTAL (BATCH*SEQ)   /* 8192 */
static constexpr size_t QKV_ELEMS = (size_t)BATCH*NHEAD*SEQ*HDIM; /* 6291456 */

// --------------------------------------------------------------------------
// Async global->LDS copy (CDNA5 GLOBAL_LOAD_ASYNC_TO_LDS_B128, ASYNCcnt).
// Builtin wants (addrspace(1) v4i*, addrspace(3) v4i*, imm offset, imm cpol).
// --------------------------------------------------------------------------
#define AS_GLOBAL __attribute__((address_space(1)))
#define AS_LDS    __attribute__((address_space(3)))
#if defined(__has_builtin)
#  if __has_builtin(__builtin_amdgcn_global_load_async_to_lds_b128)
#    define HAVE_ASYNC_LDS 1
#  endif
#endif
#ifdef HAVE_ASYNC_LDS
#  define ASYNC_CP_B128(gp, lp)                                            \
     __builtin_amdgcn_global_load_async_to_lds_b128(                       \
         (AS_GLOBAL v4i*)(gp), (AS_LDS v4i*)(lp), 0, 0)
#  if __has_builtin(__builtin_amdgcn_s_wait_asynccnt)
#    define ASYNC_WAIT() __builtin_amdgcn_s_wait_asynccnt(0)
#  else
#    define ASYNC_WAIT() asm volatile("s_wait_asynccnt 0" ::: "memory")
#  endif
#else
#  define ASYNC_CP_B128(gp, lp) (*(U128*)(lp) = *(const U128*)(gp))
#  define ASYNC_WAIT() ((void)0)
#endif

__device__ __forceinline__ u16 f32_bf16(float f) {
  unsigned int u = __float_as_uint(f);
  unsigned int r = (u + 0x7FFFu + ((u >> 16) & 1u)) >> 16;   // RNE
  return (u16)r;
}

__device__ __forceinline__ v8f zero_v8f() {
  v8f z;
#pragma unroll
  for (int i = 0; i < 8; i++) z[i] = 0.0f;
  return z;
}

__device__ __forceinline__ v8f wmma_bf16(v16bf a, v16bf b, v8f c) {
  // (neg_a, A, neg_b, B, c_mod, C, reuse_a, reuse_b)
  return __builtin_amdgcn_wmma_f32_16x16x32_bf16(false, a, false, b, (short)0, c,
                                                 false, false);
}

// --------------------------------------------------------------------------
// Pure-VALU 16-lane reductions via DPP (no LDS bpermute round-trips).
// Steps: xor1, xor2 (quad_perm), then half_mirror/mirror which act as
// xor4/xor8 once quads/octets are uniform. DPP rows are 16 lanes, so the
// two half-waves reduce independently (matching the WMMA C/D row layout).
// --------------------------------------------------------------------------
template <int CTRL>
__device__ __forceinline__ float dpp_f32(float x) {
  int i = __builtin_amdgcn_update_dpp(0, __float_as_int(x), CTRL, 0xf, 0xf, true);
  return __int_as_float(i);
}
__device__ __forceinline__ float rowmax16(float x) {
  x = fmaxf(x, dpp_f32<0xB1>(x));   // quad_perm(1,0,3,2)
  x = fmaxf(x, dpp_f32<0x4E>(x));   // quad_perm(2,3,0,1)
  x = fmaxf(x, dpp_f32<0x141>(x));  // row_half_mirror
  x = fmaxf(x, dpp_f32<0x140>(x));  // row_mirror
  return x;
}
__device__ __forceinline__ float rowsum16(float x) {
  x += dpp_f32<0xB1>(x);
  x += dpp_f32<0x4E>(x);
  x += dpp_f32<0x141>(x);
  x += dpp_f32<0x140>(x);
  return x;
}

// ---------------------------------------------------------------------------
// fp32 -> bf16 convert (grid-stride)
// ---------------------------------------------------------------------------
__global__ void cvt_f32_bf16(const float* __restrict__ in, u16* __restrict__ out,
                             int n) {
  int i = blockIdx.x * blockDim.x + threadIdx.x;
  int stride = gridDim.x * blockDim.x;
  for (; i < n; i += stride) out[i] = f32_bf16(in[i]);
}

// ---------------------------------------------------------------------------
// bf16 GEMM: C[M x NDIM] = A[M x 768] * B[768 x NDIM] + bias
//   EPI==0: QKV epilogue -> scatter bf16 into q/k/v [B,H,N,D] buffers
//   EPI==1: fp32 output row-major (+ bias)
// Block: 128x128 tile, 8 waves (4 along M x 2 along N), each wave 32x64.
// ---------------------------------------------------------------------------
template <int NDIM, int EPI>
__global__ __launch_bounds__(256) void gemm_bf16_kernel(
    const u16* __restrict__ A, const u16* __restrict__ Bm,
    const float* __restrict__ bias, u16* __restrict__ qkv_out,
    float* __restrict__ f32_out) {
  constexpr int K = CDIM, BM = 128, BN = 128, BK = 64;
  __shared__ u16 sA[BM * BK];    // [m][k]  16KB
  __shared__ u16 sBt[BN * BK];   // [n][k]  16KB (B transposed)

  const int tid = threadIdx.x, lane = tid & 31, wid = tid >> 5;
  const int m0 = blockIdx.y * BM, n0 = blockIdx.x * BN;
  const int moff = (wid >> 1) * 32, noff = (wid & 1) * 64;
  const int nl = lane & 15, hh = lane >> 4;

  v8f acc[2][4];
#pragma unroll
  for (int mi = 0; mi < 2; mi++)
#pragma unroll
    for (int ni = 0; ni < 4; ni++) acc[mi][ni] = zero_v8f();

  for (int k0 = 0; k0 < K; k0 += BK) {
    __syncthreads();
    // A tile: straight copy -> async global->LDS (ASYNCcnt path)
#pragma unroll
    for (int s = tid; s < BM * BK / 8; s += 256) {
      int m = s >> 3, q8 = s & 7;
      ASYNC_CP_B128(A + (size_t)(m0 + m) * K + k0 + q8 * 8,
                    sA + m * BK + q8 * 8);
    }
    // B tile transposed: [k][n] global -> [n][k] LDS (manual scatter)
#pragma unroll
    for (int s = tid; s < BK * BN / 8; s += 256) {
      int kr = s >> 4, nq = s & 15;
      V128 t;
      t.u = *(const U128*)(Bm + (size_t)(k0 + kr) * NDIM + n0 + nq * 8);
#pragma unroll
      for (int i = 0; i < 8; i++) sBt[(nq * 8 + i) * BK + kr] = t.s[i];
    }
    // Prefetch next K-step tiles into cache while we compute.
    if (k0 + BK < K) {
      __builtin_prefetch(A + (size_t)(m0 + (tid >> 1)) * K + (k0 + BK) +
                             (tid & 1) * 32, 0, 0);
      __builtin_prefetch(Bm + (size_t)(k0 + BK + (tid >> 2)) * NDIM + n0 +
                             (tid & 3) * 32, 0, 0);
    }
    ASYNC_WAIT();
    __syncthreads();

#pragma unroll
    for (int kc = 0; kc < 2; kc++) {
      const int ac0 = kc * 32 + 8 * hh;     // A frag: k = 2v+8h (+16 for v>=4)
      FragAB a0, a1;
      a0.q[0] = *(const U128*)(sA + (moff + nl) * BK + ac0);
      a0.q[1] = *(const U128*)(sA + (moff + nl) * BK + ac0 + 16);
      a1.q[0] = *(const U128*)(sA + (moff + 16 + nl) * BK + ac0);
      a1.q[1] = *(const U128*)(sA + (moff + 16 + nl) * BK + ac0 + 16);
      const int bc0 = kc * 32 + 8 * hh;     // B frag: k = 2v+8h (+16 for v>=4)
#pragma unroll
      for (int ni = 0; ni < 4; ni++) {
        FragAB b;
        b.q[0] = *(const U128*)(sBt + (noff + ni * 16 + nl) * BK + bc0);
        b.q[1] = *(const U128*)(sBt + (noff + ni * 16 + nl) * BK + bc0 + 16);
        acc[0][ni] = wmma_bf16(a0.v, b.v, acc[0][ni]);
        acc[1][ni] = wmma_bf16(a1.v, b.v, acc[1][ni]);
      }
    }
  }

  if constexpr (EPI == 0) {
    // Column block (64 wide, 64-aligned) lies inside exactly one (s, head).
    const int cw0 = n0 + noff;
    const int s_idx = cw0 / CDIM;
    const int head = (cw0 % CDIM) / HDIM;
    const int b_idx = m0 / SEQ;
    const int rbase = (m0 % SEQ) + moff;
    u16* ob = qkv_out + (size_t)s_idx * QKV_ELEMS +
              (size_t)(b_idx * NHEAD + head) * SEQ * HDIM;
#pragma unroll
    for (int mi = 0; mi < 2; mi++)
#pragma unroll
      for (int ni = 0; ni < 4; ni++)
#pragma unroll
        for (int r = 0; r < 8; r++) {
          int row = rbase + mi * 16 + r + 8 * hh;
          int d = ni * 16 + nl;
          float val = acc[mi][ni][r] + bias[cw0 + ni * 16 + nl];
          ob[(size_t)row * HDIM + d] = f32_bf16(val);
        }
  } else {
#pragma unroll
    for (int mi = 0; mi < 2; mi++)
#pragma unroll
      for (int ni = 0; ni < 4; ni++)
#pragma unroll
        for (int r = 0; r < 8; r++) {
          int row = m0 + moff + mi * 16 + r + 8 * hh;
          int col = n0 + noff + ni * 16 + nl;
          f32_out[(size_t)row * NDIM + col] = acc[mi][ni][r] + bias[col];
        }
  }
}

// ---------------------------------------------------------------------------
// Flash attention: block = (b,h) x 128 query rows; wave = 16 query rows.
// KV streamed in 32-key blocks. S = Q K^T via 4 WMMAs, online softmax (DPP
// reductions), P restaged through per-wave LDS into A-fragment layout,
// O += P V via 4 WMMAs.
// ---------------------------------------------------------------------------
__global__ __launch_bounds__(256) void flash_attn_kernel(
    const u16* __restrict__ Q, const u16* __restrict__ Kd,
    const u16* __restrict__ Vd, u16* __restrict__ ctx) {
  __shared__ u16 sK[32 * 64];       // [j][d]   4KB
  __shared__ u16 sVt[64 * 32];      // [d][j]   4KB
  __shared__ u16 sP[8][16 * 32];    // per-wave P staging, 8KB

  const int tid = threadIdx.x, lane = tid & 31, wid = tid >> 5;
  const int nl = lane & 15, hh = lane >> 4;
  const int bh = blockIdx.y;
  const int b_idx = bh / NHEAD, head = bh % NHEAD;
  const int q0 = blockIdx.x * 128 + wid * 16;
  const size_t hbase = (size_t)bh * SEQ * HDIM;
  const float scale = 0.125f;  // 1/sqrt(64)

  // Resident Q fragments (16x64 -> two 16x32 A-fragments)
  FragAB fq[2];
#pragma unroll
  for (int c = 0; c < 2; c++) {
    const u16* p = Q + hbase + (size_t)(q0 + nl) * HDIM + c * 32 + 8 * hh;
    fq[c].q[0] = *(const U128*)(p);
    fq[c].q[1] = *(const U128*)(p + 16);
  }

  float mrow[8], lrow[8];
#pragma unroll
  for (int r = 0; r < 8; r++) { mrow[r] = -3.0e38f; lrow[r] = 0.0f; }
  v8f o[4];
#pragma unroll
  for (int t = 0; t < 4; t++) o[t] = zero_v8f();

  for (int j0 = 0; j0 < SEQ; j0 += 32) {
    __syncthreads();
    {  // cooperative KV fill: 256 threads, one 16B slot each
      int jr = tid >> 3, qd = tid & 7;
      // K tile is a straight copy: async global->LDS
      ASYNC_CP_B128(Kd + hbase + (size_t)(j0 + jr) * HDIM + qd * 8,
                    sK + jr * 64 + qd * 8);
      // V tile is transposed into LDS: manual scatter
      V128 t;
      t.u = *(const U128*)(Vd + hbase + (size_t)(j0 + jr) * HDIM + qd * 8);
#pragma unroll
      for (int i = 0; i < 8; i++) sVt[(qd * 8 + i) * 32 + jr] = t.s[i];
      if (j0 + 32 < SEQ) {
        __builtin_prefetch(Kd + hbase + (size_t)(j0 + 32 + jr) * HDIM + qd * 8,
                           0, 0);
        __builtin_prefetch(Vd + hbase + (size_t)(j0 + 32 + jr) * HDIM + qd * 8,
                           0, 0);
      }
    }
    ASYNC_WAIT();
    __syncthreads();

    // S = Q K^T  (two 16x16 tiles, contraction over d=64 as 2 x k32)
    v8f s0 = zero_v8f(), s1 = zero_v8f();
#pragma unroll
    for (int c = 0; c < 2; c++) {
      const int c0 = c * 32 + 8 * hh;
      FragAB bk0, bk1;
      bk0.q[0] = *(const U128*)(sK + nl * 64 + c0);
      bk0.q[1] = *(const U128*)(sK + nl * 64 + c0 + 16);
      bk1.q[0] = *(const U128*)(sK + (16 + nl) * 64 + c0);
      bk1.q[1] = *(const U128*)(sK + (16 + nl) * 64 + c0 + 16);
      s0 = wmma_bf16(fq[c].v, bk0.v, s0);
      s1 = wmma_bf16(fq[c].v, bk1.v, s1);
    }

    // Online softmax (rows r+8h live across the 16 lanes of each half-wave)
#pragma unroll
    for (int r = 0; r < 8; r++) {
      float a = s0[r] * scale, bb = s1[r] * scale;
      float mx = rowmax16(fmaxf(a, bb));
      float mnew = fmaxf(mrow[r], mx);
      float f = __expf(mrow[r] - mnew);
      float p0 = __expf(a - mnew), p1 = __expf(bb - mnew);
      sP[wid][(r + 8 * hh) * 32 + nl] = f32_bf16(p0);
      sP[wid][(r + 8 * hh) * 32 + 16 + nl] = f32_bf16(p1);
      float rs = rowsum16(p0 + p1);
      lrow[r] = lrow[r] * f + rs;
      mrow[r] = mnew;
#pragma unroll
      for (int t = 0; t < 4; t++) o[t][r] *= f;
    }

    // P is in per-wave LDS; same-wave DS ops are in-order, wait for stores.
    asm volatile("s_wait_dscnt 0" ::: "memory");
    FragAB fp;
    fp.q[0] = *(const U128*)(&sP[wid][nl * 32 + 8 * hh]);
    fp.q[1] = *(const U128*)(&sP[wid][nl * 32 + 16 + 8 * hh]);
#pragma unroll
    for (int t = 0; t < 4; t++) {
      FragAB fv;
      fv.q[0] = *(const U128*)(sVt + (t * 16 + nl) * 32 + 8 * hh);
      fv.q[1] = *(const U128*)(sVt + (t * 16 + nl) * 32 + 8 * hh + 16);
      o[t] = wmma_bf16(fp.v, fv.v, o[t]);
    }
  }

  // Normalize and write context in [B*N, 768] bf16 (ready for proj GEMM)
#pragma unroll
  for (int r = 0; r < 8; r++) {
    float inv = 1.0f / lrow[r];
    int row = q0 + r + 8 * hh;
    u16* op = ctx + (size_t)(b_idx * SEQ + row) * CDIM + head * HDIM;
#pragma unroll
    for (int t = 0; t < 4; t++) op[t * 16 + nl] = f32_bf16(o[t][r] * inv);
  }
}

// ---------------------------------------------------------------------------
extern "C" void kernel_launch(void* const* d_in, const int* in_sizes, int n_in,
                              void* d_out, int out_size, void* d_ws,
                              size_t ws_size, hipStream_t stream) {
  (void)in_sizes; (void)n_in; (void)out_size; (void)ws_size;
  const float* x      = (const float*)d_in[0];
  const float* w_qkv  = (const float*)d_in[1];
  const float* b_qkv  = (const float*)d_in[2];
  const float* w_proj = (const float*)d_in[3];
  const float* b_proj = (const float*)d_in[4];
  float* out = (float*)d_out;

  u16* ws     = (u16*)d_ws;
  u16* xb     = ws;                                  // 8192*768
  u16* wqkvb  = xb + (size_t)MTOTAL * CDIM;          // 768*2304
  u16* wprojb = wqkvb + (size_t)CDIM * 3 * CDIM;     // 768*768
  u16* qkvb   = wprojb + (size_t)CDIM * CDIM;        // 3 * QKV_ELEMS
  u16* ctx    = qkvb + 3 * QKV_ELEMS;                // 8192*768

  cvt_f32_bf16<<<2048, 256, 0, stream>>>(x, xb, MTOTAL * CDIM);
  cvt_f32_bf16<<<2048, 256, 0, stream>>>(w_qkv, wqkvb, CDIM * 3 * CDIM);
  cvt_f32_bf16<<<1024, 256, 0, stream>>>(w_proj, wprojb, CDIM * CDIM);

  gemm_bf16_kernel<3 * CDIM, 0>
      <<<dim3(3 * CDIM / 128, MTOTAL / 128), 256, 0, stream>>>(
          xb, wqkvb, b_qkv, qkvb, nullptr);

  flash_attn_kernel<<<dim3(SEQ / 128, BATCH * NHEAD), 256, 0, stream>>>(
      qkvb, qkvb + QKV_ELEMS, qkvb + 2 * QKV_ELEMS, ctx);

  gemm_bf16_kernel<CDIM, 1>
      <<<dim3(CDIM / 128, MTOTAL / 128), 256, 0, stream>>>(
          ctx, wprojb, b_proj, nullptr, out);
}
